// GCNNalpha_39402029973528
// MI455X (gfx1250) — matrix operand order repaired and checked by reading, not compiled, not run on hardware
//
#include <hip/hip_runtime.h>
#include <hip/hip_bf16.h>

#define BGR   64
#define NNODE 512
#define FIN   128
#define FHID  128
#define FOUT  64

typedef __attribute__((ext_vector_type(16))) _Float16 v16h;
typedef __attribute__((ext_vector_type(8)))  _Float16 v8h;
typedef __attribute__((ext_vector_type(8)))  float    v8f;

// ---------------- WMMA fragment loaders (gfx1250 wave32 layouts) ----------------
// A (16x32 f16, row-major [M][K]): lane l holds row m = l&15; halves:
//   h=0..7  -> K = k0 + (l>>4)*8 + h
//   h=8..15 -> K = k0 + 16 + (l>>4)*8 + (h-8)
__device__ __forceinline__ v16h load_a_frag(const _Float16* __restrict__ A, int lda,
                                            int m0, int k0, int lane) {
  const int m = m0 + (lane & 15);
  const int g = lane >> 4;
  const _Float16* p = A + (size_t)m * lda + k0 + g * 8;
  v8h lo = *(const v8h*)(p);
  v8h hi = *(const v8h*)(p + 16);
  v16h a;
#pragma unroll
  for (int i = 0; i < 8; ++i) { a[i] = lo[i]; a[i + 8] = hi[i]; }
  return a;
}

// B (32x16 f16): lane l holds column n = l&15; halves h=0..7 -> K = k0 + (l>>4)*8 + h, h=8..15 -> K = k0 + 16 + (l>>4)*8 + (h-8).
// B is stored transposed, row-major [N][K], so each run of 8 halves is contiguous.
__device__ __forceinline__ v16h load_b_frag(const _Float16* __restrict__ Bt, int ldb,
                                            int n0, int k0, int lane) {
  const int nn = n0 + (lane & 15);
  const int g = lane >> 4;
  const _Float16* p = Bt + (size_t)nn * ldb + k0 + g * 8;
  v8h lo = *(const v8h*)(p);
  v8h hi = *(const v8h*)(p + 16);
  v16h b;
#pragma unroll
  for (int i = 0; i < 8; ++i) { b[i] = lo[i]; b[i + 8] = hi[i]; }
  return b;
}

// ---------------- Aggregation GEMM: C[g] = S[g] @ Z[g]  (per-graph 512x512 @ 512x128) ----
// S: [B][512][512] f16 row-major. Zt: [B][128][512] f16 (transposed/feature-major).
// Cnm: [B][512][128] node-major.  Ct: [B][128][512] feature-major (optional).
__global__ void __launch_bounds__(128)
agg_gemm_kernel(const _Float16* __restrict__ S, const _Float16* __restrict__ Zt,
                _Float16* __restrict__ Cnm, _Float16* __restrict__ Ct) {
  const int lane = threadIdx.x & 31;
  const int wave = threadIdx.x >> 5;
  const int g    = blockIdx.z;
  const int m0   = blockIdx.x * 256 + wave * 64;
  const int n0   = blockIdx.y * 64;

  const _Float16* A  = S  + (size_t)g * NNODE * NNODE;
  const _Float16* Bt = Zt + (size_t)g * FHID * NNODE;

  v8f c[4][4] = {};
  for (int k0 = 0; k0 < NNODE; k0 += 32) {
    v16h af[4], bf[4];
#pragma unroll
    for (int i = 0; i < 4; ++i) af[i] = load_a_frag(A, NNODE, m0 + 16 * i, k0, lane);
#pragma unroll
    for (int j = 0; j < 4; ++j) bf[j] = load_b_frag(Bt, NNODE, n0 + 16 * j, k0, lane);
#pragma unroll
    for (int i = 0; i < 4; ++i)
#pragma unroll
      for (int j = 0; j < 4; ++j)
        c[i][j] = __builtin_amdgcn_wmma_f32_16x16x32_f16(
            false, af[i], false, bf[j], (short)0, c[i][j], false, false);
  }

  const int ncol = lane & 15;
  const int gh   = lane >> 4;
  _Float16* cbase = Cnm + (size_t)g * NNODE * FHID;
#pragma unroll
  for (int i = 0; i < 4; ++i) {
#pragma unroll
    for (int j = 0; j < 4; ++j) {
      const int mt = m0 + 16 * i;
      const int nt = n0 + 16 * j + ncol;
      v8h hv;
#pragma unroll
      for (int v = 0; v < 8; ++v) hv[v] = (_Float16)c[i][j][v];
#pragma unroll
      for (int v = 0; v < 8; ++v)
        cbase[(size_t)(mt + v + 8 * gh) * FHID + nt] = hv[v];
      if (Ct) {
        _Float16* tp = Ct + ((size_t)g * FHID + nt) * NNODE + mt + 8 * gh;
        *(v8h*)tp = hv;
      }
    }
  }
}

// ---------------- Feature GEMM: C = ReLU(A0@W0 + A1@W1 + A2@W2 + b) --------------------
// A_t: [M=32768][128] node-major f16. Wt: [3][128][128] f16, stored [tap][out][in].
__global__ void __launch_bounds__(128)
feat_gemm_kernel(const _Float16* __restrict__ A0, const _Float16* __restrict__ A1,
                 const _Float16* __restrict__ A2, const _Float16* __restrict__ Wt,
                 const float* __restrict__ bias,
                 _Float16* __restrict__ Cnm, _Float16* __restrict__ Ct) {
  const int lane = threadIdx.x & 31;
  const int wave = threadIdx.x >> 5;
  const int m0   = blockIdx.x * 256 + wave * 64;
  const int n0   = blockIdx.y * 64;

  const _Float16* As[3] = {A0, A1, A2};
  v8f c[4][4] = {};
#pragma unroll
  for (int t = 0; t < 3; ++t) {
    const _Float16* Bt = Wt + (size_t)t * FHID * FHID;
    for (int k0 = 0; k0 < FHID; k0 += 32) {
      v16h af[4], bf[4];
#pragma unroll
      for (int i = 0; i < 4; ++i) af[i] = load_a_frag(As[t], FHID, m0 + 16 * i, k0, lane);
#pragma unroll
      for (int j = 0; j < 4; ++j) bf[j] = load_b_frag(Bt, FHID, n0 + 16 * j, k0, lane);
#pragma unroll
      for (int i = 0; i < 4; ++i)
#pragma unroll
        for (int j = 0; j < 4; ++j)
          c[i][j] = __builtin_amdgcn_wmma_f32_16x16x32_f16(
              false, af[i], false, bf[j], (short)0, c[i][j], false, false);
    }
  }

  const int ncol = lane & 15;
  const int gh   = lane >> 4;
#pragma unroll
  for (int i = 0; i < 4; ++i) {
#pragma unroll
    for (int j = 0; j < 4; ++j) {
      const int mt = m0 + 16 * i;
      const int nt = n0 + 16 * j + ncol;
      const float bb = bias[nt];
      v8h hv;
#pragma unroll
      for (int v = 0; v < 8; ++v) {
        float x = c[i][j][v] + bb;
        hv[v] = (_Float16)(x > 0.0f ? x : 0.0f);
      }
#pragma unroll
      for (int v = 0; v < 8; ++v)
        Cnm[(size_t)(mt + v + 8 * gh) * FHID + nt] = hv[v];
      if (Ct) {
        const int gg   = mt / NNODE;      // whole tile lies in one graph (512 % 64 == 0)
        const int mloc = mt % NNODE;
        _Float16* tp = Ct + ((size_t)gg * FHID + nt) * NNODE + mloc + 8 * gh;
        *(v8h*)tp = hv;
      }
    }
  }
}

// ---------------- Setup / small kernels -------------------------------------------------
__global__ void zero_f4_kernel(float4* __restrict__ p, int n4) {
  int i = blockIdx.x * blockDim.x + threadIdx.x;
  if (i < n4) p[i] = make_float4(0.f, 0.f, 0.f, 0.f);
}

__global__ void scatter_kernel(const int* __restrict__ ei, int E, float* __restrict__ S32) {
  int i = blockIdx.x * blockDim.x + threadIdx.x;
  if (i >= E) return;
  int s = ei[i];
  int d = ei[E + i];
  int g = s >> 9;            // / 512
  int r = s & (NNODE - 1);
  int cc = d & (NNODE - 1);
  atomicAdd(&S32[((size_t)g * NNODE + r) * NNODE + cc], 1.0f);
}

__global__ void cvt_s_kernel(const float* __restrict__ S32, const float* __restrict__ alpha_p,
                             _Float16* __restrict__ S16) {
  size_t i = (size_t)blockIdx.x * blockDim.x + threadIdx.x;
  const size_t total = (size_t)BGR * NNODE * NNODE;
  if (i >= total) return;
  const float alpha = alpha_p[0];
  int rc = (int)(i & (NNODE * NNODE - 1));
  int r = rc >> 9;
  int cc = rc & (NNODE - 1);
  float v = alpha * S32[i] + ((r == cc) ? (1.0f - alpha) : 0.0f);
  S16[i] = (_Float16)v;
}

__global__ void cvt_x_kernel(const float* __restrict__ X, _Float16* __restrict__ Xnm,
                             _Float16* __restrict__ Xt) {
  size_t i = (size_t)blockIdx.x * blockDim.x + threadIdx.x;
  const size_t total = (size_t)BGR * NNODE * FIN;
  if (i >= total) return;
  int m = (int)(i >> 7);
  int f = (int)(i & 127);
  _Float16 v = (_Float16)X[i];
  Xnm[i] = v;
  Xt[((size_t)(m >> 9) * FHID + f) * NNODE + (m & (NNODE - 1))] = v;
}

__global__ void cvt_w_kernel(const float* __restrict__ W, _Float16* __restrict__ Wt, int taps) {
  int i = blockIdx.x * blockDim.x + threadIdx.x;
  if (i >= taps * FHID * FHID) return;
  int t = i / (FHID * FHID);
  int rem = i - t * FHID * FHID;
  int fin = rem >> 7;
  int fout = rem & 127;
  Wt[(size_t)t * FHID * FHID + fout * FHID + fin] = (_Float16)W[i];
}

__global__ void pool_kernel(const _Float16* __restrict__ H, float* __restrict__ pooled) {
  int b = blockIdx.x;
  int f = threadIdx.x;                 // blockDim.x == 128
  float m = -3.0e38f;
  for (int node = 0; node < NNODE; ++node) {
    float v = (float)H[((size_t)b * NNODE + node) * FHID + f];
    m = fmaxf(m, v);
  }
  pooled[b * FHID + f] = m;
}

__global__ void out_kernel(const float* __restrict__ pooled, const float* __restrict__ Wout,
                           const float* __restrict__ bout, float* __restrict__ out) {
  int b = blockIdx.x;
  int o = threadIdx.x;                 // blockDim.x == 64
  float acc = bout[o];
  for (int f = 0; f < FHID; ++f)
    acc += pooled[b * FHID + f] * Wout[f * FOUT + o];
  out[b * FOUT + o] = acc;
}

// ---------------- Host side -------------------------------------------------------------
extern "C" void kernel_launch(void* const* d_in, const int* in_sizes, int n_in,
                              void* d_out, int out_size, void* d_ws, size_t ws_size,
                              hipStream_t stream) {
  const float* X      = (const float*)d_in[0];
  const int*   edges  = (const int*)d_in[2];
  const float* W1     = (const float*)d_in[3];
  const float* b1     = (const float*)d_in[4];
  const float* W2     = (const float*)d_in[5];
  const float* b2     = (const float*)d_in[6];
  const float* Wout   = (const float*)d_in[7];
  const float* bout   = (const float*)d_in[8];
  const float* alpha  = (const float*)d_in[9];
  float* out          = (float*)d_out;

  const int E = in_sizes[2] / 2;

  // workspace layout (bytes)
  char* ws = (char*)d_ws;
  size_t off = 0;
  auto take = [&](size_t bytes) { char* p = ws + off; off += (bytes + 255) & ~(size_t)255; return p; };
  const size_t featB = (size_t)BGR * NNODE * FHID * sizeof(_Float16);   // 8 MB
  _Float16* S16  = (_Float16*)take((size_t)BGR * NNODE * NNODE * sizeof(_Float16)); // 32 MB
  _Float16* A0nm = (_Float16*)take(featB);   // X f16 node-major; reused for H2
  _Float16* A0t  = (_Float16*)take(featB);   // X f16 feature-major
  _Float16* Z1nm = (_Float16*)take(featB);
  _Float16* Z1t  = (_Float16*)take(featB);
  _Float16* Z2nm = (_Float16*)take(featB);
  _Float16* H1nm = (_Float16*)take(featB);
  _Float16* H1t  = (_Float16*)take(featB);
  _Float16* Wt1  = (_Float16*)take(3 * FHID * FHID * sizeof(_Float16));
  _Float16* Wt2  = (_Float16*)take(3 * FHID * FHID * sizeof(_Float16));
  float*    pooled = (float*)take((size_t)BGR * FHID * sizeof(float));
  float*    S32  = (float*)take((size_t)BGR * NNODE * NNODE * sizeof(float)); // 64 MB
  (void)ws_size;

  // 1. zero the adjacency accumulator
  {
    int n4 = BGR * NNODE * NNODE / 4;
    zero_f4_kernel<<<(n4 + 255) / 256, 256, 0, stream>>>((float4*)S32, n4);
  }
  // 2. scatter edges
  scatter_kernel<<<(E + 255) / 256, 256, 0, stream>>>(edges, E, S32);
  // 3. S16 = alpha*adj + (1-alpha)*I
  {
    size_t total = (size_t)BGR * NNODE * NNODE;
    cvt_s_kernel<<<(unsigned)((total + 255) / 256), 256, 0, stream>>>(S32, alpha, S16);
  }
  // 4. convert X, weights to f16 (both layouts / transposed)
  {
    size_t total = (size_t)BGR * NNODE * FIN;
    cvt_x_kernel<<<(unsigned)((total + 255) / 256), 256, 0, stream>>>(X, A0nm, A0t);
  }
  cvt_w_kernel<<<(3 * FHID * FHID + 255) / 256, 256, 0, stream>>>(W1, Wt1, 3);
  cvt_w_kernel<<<(3 * FHID * FHID + 255) / 256, 256, 0, stream>>>(W2, Wt2, 3);

  dim3 aggGrid(NNODE / 256, FHID / 64, BGR);        // (2, 2, 64)
  dim3 featGrid(BGR * NNODE / 256, FHID / 64, 1);   // (128, 2)

  // layer 1
  agg_gemm_kernel<<<aggGrid, 128, 0, stream>>>(S16, A0t, Z1nm, Z1t);
  agg_gemm_kernel<<<aggGrid, 128, 0, stream>>>(S16, Z1t, Z2nm, nullptr);
  feat_gemm_kernel<<<featGrid, 128, 0, stream>>>(A0nm, Z1nm, Z2nm, Wt1, b1, H1nm, H1t);
  // layer 2
  agg_gemm_kernel<<<aggGrid, 128, 0, stream>>>(S16, H1t, Z1nm, Z1t);
  agg_gemm_kernel<<<aggGrid, 128, 0, stream>>>(S16, Z1t, Z2nm, nullptr);
  feat_gemm_kernel<<<featGrid, 128, 0, stream>>>(H1nm, Z1nm, Z2nm, Wt2, b2, A0nm, nullptr);
  // pool + output projection
  pool_kernel<<<BGR, FHID, 0, stream>>>(A0nm, pooled);
  out_kernel<<<BGR, FOUT, 0, stream>>>(pooled, Wout, bout, out);
  (void)n_in; (void)out_size;
}
